// CustomMHA_71760313581643
// MI455X (gfx1250) — matrix operand order repaired and checked by reading, not compiled, not run on hardware
//
#include <hip/hip_runtime.h>
#include <hip/hip_bf16.h>

typedef __attribute__((ext_vector_type(16))) __bf16 v16bf;
typedef __attribute__((ext_vector_type(8)))  __bf16 v8bf;
typedef __attribute__((ext_vector_type(8)))  float  v8f;
typedef __attribute__((ext_vector_type(4)))  unsigned int v4u;
typedef __attribute__((ext_vector_type(8)))  int v8i_;
typedef __attribute__((ext_vector_type(4)))  int v4i_;

#define WMMA_BF16(A,B,C) \
  __builtin_amdgcn_wmma_f32_16x16x32_bf16(false,(A),false,(B),(short)0,(C),false,false)

// CDNA5 async global->LDS copy (ASYNCcnt-tracked), ISA 15.18.3 op 98.
// ldsoff: wave-relative LDS byte address (low 32 bits of generic LDS pointer,
// per the aperture mapping); gaddr: 64-bit global addr.
#if defined(__AMDGCN__)
#define ASYNC_LDS_B128(ldsoff, gaddr)                                   \
  asm volatile("global_load_async_to_lds_b128 %0, %1, off"              \
               :: "v"(ldsoff), "v"(gaddr) : "memory")
#define WAIT_ASYNC() asm volatile("s_wait_asynccnt 0x0" ::: "memory")
#else
#define ASYNC_LDS_B128(ldsoff, gaddr) ((void)(ldsoff), (void)(gaddr))
#define WAIT_ASYNC() ((void)0)
#endif

// CDNA5 Tensor Data Mover: wave-level DMA of a contiguous 1-D tile to LDS.
// D# per ISA 8.3/8.4: group0 = {count=1, lds_addr, global_addr[56:0], type=2},
// group1 = {mask=0, data_size=2B, tensor_dim0=tile_dim0=nelem, dim1=1}.
// This toolchain exposes the 6-arg builtin: (g0, g1, g2, g3, g1b, cpol).
#if defined(__AMDGCN__) && __has_builtin(__builtin_amdgcn_tensor_load_to_lds) && \
    __has_builtin(__builtin_amdgcn_s_wait_tensorcnt)
#define HAVE_TDM 1
static __device__ __forceinline__ void tdm_load_1d(unsigned lds_addr,
                                                   const void* gptr,
                                                   unsigned nelem) {
    const unsigned long long ga = (unsigned long long)(uintptr_t)gptr;
    v4u g0;
    g0[0] = 1u;                                    // count=1, user descriptor
    g0[1] = lds_addr;                              // LDS byte address
    g0[2] = (unsigned)ga;                          // global_addr[31:0]
    g0[3] = (unsigned)((ga >> 32) & 0x01FFFFFFu)   // global_addr[56:32]
          | 0x80000000u;                           // type=2 ("image")
    v8i_ g1;
    g1[0] = 1 << 16;                               // data_size = 2 bytes
    g1[1] = (int)((nelem & 0xFFFFu) << 16);        // tensor_dim0[15:0]
    g1[2] = (int)((nelem >> 16) & 0xFFFFu)         // tensor_dim0[31:16]
          | (1 << 16);                             // tensor_dim1 = 1
    g1[3] = (int)((nelem & 0xFFFFu) << 16);        // tile_dim0 = nelem
    g1[4] = 0;                                     // tile_dim1/2 = 0 (1-D)
    g1[5] = (int)nelem;                            // tensor_dim0_stride
    g1[6] = 0;
    g1[7] = 0;
    const v4i_ z4 = {0, 0, 0, 0};
    const v8i_ z8 = {0, 0, 0, 0, 0, 0, 0, 0};
    __builtin_amdgcn_tensor_load_to_lds(g0, g1, z4, z4, z8, 0);
}
#define WAIT_TENSOR() __builtin_amdgcn_s_wait_tensorcnt(0)
#else
#define HAVE_TDM 0
#define WAIT_TENSOR() ((void)0)
#endif

static __device__ __forceinline__ void splitf(float x, __bf16& h, __bf16& l) {
    h = (__bf16)x;
    l = (__bf16)(x - (float)h);
}

// ---------------------------------------------------------------------------
// fp32 -> (hi, lo) bf16 split
// ---------------------------------------------------------------------------
__global__ void split_kernel(const float* __restrict__ in,
                             __bf16* __restrict__ hi,
                             __bf16* __restrict__ lo, int n) {
    int i = blockIdx.x * blockDim.x + threadIdx.x;
    if (i < n) {
        __bf16 h, l;
        splitf(in[i], h, l);
        hi[i] = h; lo[i] = l;
    }
}

// A-frag k order used here (contiguous, matches B): lane<16 -> K {0-15}, lane>=16 -> K {16-31}
static __device__ __forceinline__ v16bf load_afrag(const __bf16* rowp, int o) {
    v8bf g0 = *(const v8bf*)(rowp + o);
    v8bf g1 = *(const v8bf*)(rowp + o + 8);
    v16bf v;
    for (int e = 0; e < 8; ++e) { v[e] = g0[e]; v[8 + e] = g1[e]; }
    return v;
}

// ---------------------------------------------------------------------------
// 32x32 output macro-tile (2x2 WMMA tiles) of C = A * B^T,
// bf16x3 split-precision with fp32 accumulate. A: MxK row-major, B: NxK
// row-major; Ah/Al and Bh/Bl point at this macro-tile's first row.
// B-frag k order used here (contiguous, matches A): lane<16 -> K {0-15}, lane>=16 -> K {16-31}
// ---------------------------------------------------------------------------
static __device__ __forceinline__ void gemm_2x2_bf16x3(
    const __bf16* __restrict__ Ah, const __bf16* __restrict__ Al,
    const __bf16* __restrict__ Bh, const __bf16* __restrict__ Bl,
    int K, int lane, v8f acc[2][2])
{
    const int r    = lane & 15;
    const int half = lane >> 4;
    const __bf16 *arh[2], *arl[2], *brh[2], *brl[2];
    for (int t = 0; t < 2; ++t) {
        arh[t] = Ah + (size_t)(t * 16 + r) * K;
        arl[t] = Al + (size_t)(t * 16 + r) * K;
        brh[t] = Bh + (size_t)(t * 16 + r) * K;
        brl[t] = Bl + (size_t)(t * 16 + r) * K;
    }
    for (int kk = 0; kk < K; kk += 32) {
        const int ao = kk + (half ? 16 : 0);
        const int bo = kk + (half ? 16 : 0);
        v16bf ah[2], al[2], bh[2], bl[2];
        for (int t = 0; t < 2; ++t) {
            ah[t] = load_afrag(arh[t], ao);
            al[t] = load_afrag(arl[t], ao);
            bh[t] = *(const v16bf*)(brh[t] + bo);
            bl[t] = *(const v16bf*)(brl[t] + bo);
        }
        for (int mt = 0; mt < 2; ++mt)
            for (int nt = 0; nt < 2; ++nt) {
                acc[mt][nt] = WMMA_BF16(ah[mt], bh[nt], acc[mt][nt]);
                acc[mt][nt] = WMMA_BF16(ah[mt], bl[nt], acc[mt][nt]);
                acc[mt][nt] = WMMA_BF16(al[mt], bh[nt], acc[mt][nt]);
            }
    }
}

// ---------------------------------------------------------------------------
// QKV projection: t = x @ W_qkv^T, M=4096 N=3072 K=1024.
// Writes Q/K/V as (B,S,D) hi/lo bf16 buffers (columns 0-1023 -> Q, etc).
// ---------------------------------------------------------------------------
__global__ void __launch_bounds__(256) qkv_gemm_kernel(
    const __bf16* __restrict__ xh, const __bf16* __restrict__ xl,
    const __bf16* __restrict__ wh, const __bf16* __restrict__ wl,
    __bf16* __restrict__ Qh, __bf16* __restrict__ Ql,
    __bf16* __restrict__ Kh, __bf16* __restrict__ Kl,
    __bf16* __restrict__ Vh, __bf16* __restrict__ Vl)
{
    const int K = 1024, N = 3072;
    const int wave = threadIdx.x >> 5, lane = threadIdx.x & 31;
    const int gw  = blockIdx.x * 8 + wave;
    const int tn2 = gw % (N / 32);
    const int tm2 = gw / (N / 32);

    v8f acc[2][2] = {};
    gemm_2x2_bf16x3(xh + (size_t)tm2 * 32 * K, xl + (size_t)tm2 * 32 * K,
                    wh + (size_t)tn2 * 32 * K, wl + (size_t)tn2 * 32 * K,
                    K, lane, acc);

    const int r = lane & 15, half = lane >> 4;
    for (int nt = 0; nt < 2; ++nt) {
        const int col = tn2 * 32 + nt * 16 + r;
        const int sel = col >> 10;         // 0=Q 1=K 2=V
        const int cc  = col & 1023;
        __bf16 *dh, *dl;
        if (sel == 0)      { dh = Qh; dl = Ql; }
        else if (sel == 1) { dh = Kh; dl = Kl; }
        else               { dh = Vh; dl = Vl; }
        for (int mt = 0; mt < 2; ++mt)
            for (int i = 0; i < 8; ++i) {
                const int m = tm2 * 32 + mt * 16 + i + 8 * half;
                const size_t idx = (size_t)m * 1024 + cc;
                __bf16 h, l;
                splitf(acc[mt][nt][i], h, l);
                dh[idx] = h; dl[idx] = l;
            }
    }
}

// ---------------------------------------------------------------------------
// Fused causal attention, one block per (b, head, 128-row q-block).
// 8 waves; each wave owns 16 query rows. K chunk staged via TDM
// tensor_load_to_lds (TENSORcnt); V chunk staged row-major via async
// global->LDS copies (ASYNCcnt) and transposed LDS->LDS through VGPRs.
// Replicates reference quirks: scale 1/sqrt(D)=1/32, tril, score==0 -> -inf.
// ---------------------------------------------------------------------------
__global__ void __launch_bounds__(256) flash_kernel(
    const __bf16* __restrict__ Qh, const __bf16* __restrict__ Ql,
    const __bf16* __restrict__ Kh, const __bf16* __restrict__ Kl,
    const __bf16* __restrict__ Vh, const __bf16* __restrict__ Vl,
    __bf16* __restrict__ Oh, __bf16* __restrict__ Ol)
{
    const int S = 2048, DH = 64;
    __shared__ __bf16 kl_h[32 * 64], kl_l[32 * 64];     // K chunk, row-major
    __shared__ __bf16 vrow_h[32 * 64], vrow_l[32 * 64]; // V chunk, row-major
    __shared__ __bf16 vt_h[64 * 32], vt_l[64 * 32];     // V chunk, transposed
    __shared__ float  pbuf[8][16][32];                  // per-wave P transpose

    const int blk = blockIdx.x;        // B*H*(S/128) = 512
    const int qb  = blk & 15;
    const int hh  = (blk >> 4) & 15;
    const int b   = blk >> 8;
    const size_t hoff = (size_t)b * S * 1024 + (size_t)hh * S * DH;

    const int wave = threadIdx.x >> 5, lane = threadIdx.x & 31;
    const int r = lane & 15, half = lane >> 4;
    const int qr0 = qb * 128 + wave * 16;

    // q fragments (A-layout), 2 K-steps of 32 features, hi/lo
    const __bf16* qrow_h = Qh + hoff + (size_t)(qr0 + r) * DH;
    const __bf16* qrow_l = Ql + hoff + (size_t)(qr0 + r) * DH;
    v16bf qfh[2], qfl[2];
    for (int s = 0; s < 2; ++s) {
        const int ao = 32 * s + (half ? 16 : 0);
        qfh[s] = load_afrag(qrow_h, ao);
        qfl[s] = load_afrag(qrow_l, ao);
    }

    const float NEG_INF = -__builtin_inff();
    float mrow[8], lrow[8];
    v8f accv[4] = {};
    for (int i = 0; i < 8; ++i) { mrow[i] = NEG_INF; lrow[i] = 0.f; }

    const int nch = qb * 4 + 4;  // 32-col chunks covering k <= block max q
    for (int c = 0; c < nch; ++c) {
        const int kc0 = c * 32;
        __syncthreads();
        {   // cooperative stage: K via TDM (wave 0), V rows via async copies
            const int t = threadIdx.x;
            const int row = t >> 3, c8 = (t & 7) * 8;
            const size_t g = hoff + (size_t)(kc0 + row) * DH + c8;

#if HAVE_TDM
            if (wave == 0) {
                tdm_load_1d((unsigned)(uintptr_t)&kl_h[0],
                            Kh + hoff + (size_t)kc0 * DH, 32 * DH);
                tdm_load_1d((unsigned)(uintptr_t)&kl_l[0],
                            Kl + hoff + (size_t)kc0 * DH, 32 * DH);
            }
#else
            ASYNC_LDS_B128((unsigned)(uintptr_t)&kl_h[row * 64 + c8],
                           (unsigned long long)(uintptr_t)(Kh + g));
            ASYNC_LDS_B128((unsigned)(uintptr_t)&kl_l[row * 64 + c8],
                           (unsigned long long)(uintptr_t)(Kl + g));
#endif
            ASYNC_LDS_B128((unsigned)(uintptr_t)&vrow_h[row * 64 + c8],
                           (unsigned long long)(uintptr_t)(Vh + g));
            ASYNC_LDS_B128((unsigned)(uintptr_t)&vrow_l[row * 64 + c8],
                           (unsigned long long)(uintptr_t)(Vl + g));
            WAIT_ASYNC();
#if HAVE_TDM
            if (wave == 0) WAIT_TENSOR();
#endif
        }
        __syncthreads();
        {   // transpose V: LDS row-major -> LDS feature-major
            const int t = threadIdx.x;
            const int row = t >> 3, c8 = (t & 7) * 8;
            v8bf v8h = *(const v8bf*)&vrow_h[row * 64 + c8];
            v8bf v8l = *(const v8bf*)&vrow_l[row * 64 + c8];
            for (int e = 0; e < 8; ++e) {
                vt_h[(c8 + e) * 32 + row] = v8h[e];
                vt_l[(c8 + e) * 32 + row] = v8l[e];
            }
        }
        __syncthreads();

        // scores: two 16x16 tiles (chunk cols 0-15, 16-31)
        v8f st[2];
        for (int kt = 0; kt < 2; ++kt) {
            const __bf16* kr_h = &kl_h[(kt * 16 + r) * 64];
            const __bf16* kr_l = &kl_l[(kt * 16 + r) * 64];
            v8f sacc = {};
            for (int s = 0; s < 2; ++s) {
                const int bo = 32 * s + (half ? 16 : 0);
                v16bf bh = *(const v16bf*)(kr_h + bo);
                v16bf bl = *(const v16bf*)(kr_l + bo);
                sacc = WMMA_BF16(qfh[s], bh, sacc);
                sacc = WMMA_BF16(qfh[s], bl, sacc);
                sacc = WMMA_BF16(qfl[s], bh, sacc);
            }
            st[kt] = sacc;
        }

        // scale + causal/zero mask + online softmax update
        const float rscale = 0.03125f;  // 1/sqrt(1024)
        float p0[8], p1[8], alpha[8];
        for (int i = 0; i < 8; ++i) {
            const int qr = qr0 + i + 8 * half;
            float s0 = st[0][i] * rscale;
            float s1 = st[1][i] * rscale;
            if ((kc0 + r) > qr      || s0 == 0.f) s0 = NEG_INF;
            if ((kc0 + 16 + r) > qr || s1 == 0.f) s1 = NEG_INF;
            float cm = fmaxf(s0, s1);
            for (int msk = 8; msk; msk >>= 1)
                cm = fmaxf(cm, __shfl_xor(cm, msk, 32));
            const float mnew = fmaxf(mrow[i], cm);
            float a = (mrow[i] == NEG_INF) ? 0.f : __expf(mrow[i] - mnew);
            p0[i] = (s0 == NEG_INF) ? 0.f : __expf(s0 - mnew);
            p1[i] = (s1 == NEG_INF) ? 0.f : __expf(s1 - mnew);
            float rsum = p0[i] + p1[i];
            for (int msk = 8; msk; msk >>= 1)
                rsum += __shfl_xor(rsum, msk, 32);
            lrow[i]  = lrow[i] * a + rsum;
            mrow[i]  = mnew;
            alpha[i] = a;
        }
        for (int t = 0; t < 4; ++t)
            for (int i = 0; i < 8; ++i)
                accv[t][i] *= alpha[i];

        // P: C-layout -> LDS -> A-layout (per-wave scratch), split hi/lo
        for (int i = 0; i < 8; ++i) {
            pbuf[wave][i + 8 * half][r]      = p0[i];
            pbuf[wave][i + 8 * half][16 + r] = p1[i];
        }
        __builtin_amdgcn_wave_barrier();
        v16bf pa_h, pa_l;
        {
            const float* prow = &pbuf[wave][r][0];
            const int o0 = half ? 16 : 0;
            for (int e = 0; e < 8; ++e) {
                __bf16 h0, l0, h1, l1;
                splitf(prow[o0 + e],      h0, l0);
                splitf(prow[o0 + 8 + e], h1, l1);
                pa_h[e] = h0; pa_h[8 + e] = h1;
                pa_l[e] = l0; pa_l[8 + e] = l1;
            }
        }

        // out += P @ V  (4 N-tiles of 16 features)
        for (int t = 0; t < 4; ++t) {
            const __bf16* vr_h = &vt_h[(t * 16 + r) * 32 + (half ? 16 : 0)];
            const __bf16* vr_l = &vt_l[(t * 16 + r) * 32 + (half ? 16 : 0)];
            v16bf vbh = *(const v16bf*)vr_h;
            v16bf vbl = *(const v16bf*)vr_l;
            accv[t] = WMMA_BF16(pa_h, vbh, accv[t]);
            accv[t] = WMMA_BF16(pa_h, vbl, accv[t]);
            accv[t] = WMMA_BF16(pa_l, vbh, accv[t]);
        }
    }

    // epilogue: normalize, write O[b, s', h*64 + f] as hi/lo bf16
    for (int t = 0; t < 4; ++t)
        for (int i = 0; i < 8; ++i) {
            const int row = qr0 + i + 8 * half;
            const float v = accv[t][i] / lrow[i];
            __bf16 h, l;
            splitf(v, h, l);
            const size_t idx = (size_t)b * S * 1024 + (size_t)row * 1024
                             + hh * 64 + t * 16 + r;
            Oh[idx] = h; Ol[idx] = l;
        }
}

// ---------------------------------------------------------------------------
// Output projection: y = O @ W_o^T, M=4096 N=1024 K=1024, fp32 out.
// ---------------------------------------------------------------------------
__global__ void __launch_bounds__(256) out_gemm_kernel(
    const __bf16* __restrict__ oh, const __bf16* __restrict__ ol,
    const __bf16* __restrict__ wh, const __bf16* __restrict__ wl,
    float* __restrict__ y)
{
    const int K = 1024, N = 1024;
    const int wave = threadIdx.x >> 5, lane = threadIdx.x & 31;
    const int gw  = blockIdx.x * 8 + wave;
    const int tn2 = gw % (N / 32);
    const int tm2 = gw / (N / 32);

    v8f acc[2][2] = {};
    gemm_2x2_bf16x3(oh + (size_t)tm2 * 32 * K, ol + (size_t)tm2 * 32 * K,
                    wh + (size_t)tn2 * 32 * K, wl + (size_t)tn2 * 32 * K,
                    K, lane, acc);

    const int r = lane & 15, half = lane >> 4;
    for (int nt = 0; nt < 2; ++nt) {
        const int col = tn2 * 32 + nt * 16 + r;
        for (int mt = 0; mt < 2; ++mt)
            for (int i = 0; i < 8; ++i) {
                const int m = tm2 * 32 + mt * 16 + i + 8 * half;
                y[(size_t)m * 1024 + col] = acc[mt][nt][i];
            }
    }
}

// ---------------------------------------------------------------------------
extern "C" void kernel_launch(void* const* d_in, const int* in_sizes, int n_in,
                              void* d_out, int out_size, void* d_ws, size_t ws_size,
                              hipStream_t stream)
{
    (void)in_sizes; (void)n_in; (void)out_size; (void)ws_size;
    const float* x    = (const float*)d_in[0];
    const float* wqkv = (const float*)d_in[1];
    const float* wo   = (const float*)d_in[2];
    float* y = (float*)d_out;

    char* ws = (char*)d_ws;
    size_t off = 0;
    auto alloc = [&](size_t bytes) {
        void* p = ws + off;
        off += (bytes + 255) & ~(size_t)255;
        return p;
    };
    const size_t nx  = 4096ull * 1024;   // x / Q / K / V / O elements
    const size_t nw  = 3072ull * 1024;   // W_qkv elements
    const size_t nwo = 1024ull * 1024;   // W_o elements

    __bf16* xh  = (__bf16*)alloc(nx * 2);
    __bf16* xl  = (__bf16*)alloc(nx * 2);
    __bf16* wqh = (__bf16*)alloc(nw * 2);
    __bf16* wql = (__bf16*)alloc(nw * 2);
    __bf16* woh = (__bf16*)alloc(nwo * 2);
    __bf16* wol = (__bf16*)alloc(nwo * 2);
    __bf16* Qh  = (__bf16*)alloc(nx * 2);
    __bf16* Ql  = (__bf16*)alloc(nx * 2);
    __bf16* Kh  = (__bf16*)alloc(nx * 2);
    __bf16* Kl  = (__bf16*)alloc(nx * 2);
    __bf16* Vh  = (__bf16*)alloc(nx * 2);
    __bf16* Vl  = (__bf16*)alloc(nx * 2);
    __bf16* Oh  = (__bf16*)alloc(nx * 2);
    __bf16* Ol  = (__bf16*)alloc(nx * 2);

    split_kernel<<<(int)((nx  + 255) / 256), 256, 0, stream>>>(x,    xh,  xl,  (int)nx);
    split_kernel<<<(int)((nw  + 255) / 256), 256, 0, stream>>>(wqkv, wqh, wql, (int)nw);
    split_kernel<<<(int)((nwo + 255) / 256), 256, 0, stream>>>(wo,   woh, wol, (int)nwo);

    // QKV: (4096/32)*(3072/32) = 12288 wave-tiles / 8 waves per block
    qkv_gemm_kernel<<<1536, 256, 0, stream>>>(xh, xl, wqh, wql,
                                              Qh, Ql, Kh, Kl, Vh, Vl);
    flash_kernel<<<512, 256, 0, stream>>>(Qh, Ql, Kh, Kl, Vh, Vl, Oh, Ol);
    // Out-proj: (4096/32)*(1024/32) = 4096 wave-tiles / 8 per block
    out_gemm_kernel<<<512, 256, 0, stream>>>(Oh, Ol, woh, wol, y);
}
